// LinearSVDBlock_1468878815674
// MI455X (gfx1250) — compile-verified, matrix operand order checked by reading, not run on hardware
//
#include <hip/hip_runtime.h>
#include <hip/hip_bf16.h>
#include <math.h>

// ---------------- problem constants (match reference) ----------------
#define BB 2
#define SS 2048
#define DD 1024
#define HH 16
#define EE 64
#define DFF 4096
#define R_ATTN 32
#define R_OUT 512
#define R_FF 512
#define N1 (BB * SS)          // 4096 token rows
#define SCALE_QK 0.125f       // 1/sqrt(64)

typedef _Float16 half_t;
typedef __attribute__((ext_vector_type(16))) _Float16 v16h;
typedef __attribute__((ext_vector_type(8)))  float    v8f;
typedef int v4i __attribute__((vector_size(16)));
typedef __attribute__((address_space(1))) v4i* gas_v4i;
typedef __attribute__((address_space(3))) v4i* las_v4i;

// CDNA5 async global->LDS staging (ASYNCcnt) if the toolchain exposes it.
#if __has_builtin(__builtin_amdgcn_global_load_async_to_lds_b128) && \
    __has_builtin(__builtin_amdgcn_s_wait_asynccnt)
#define USE_ASYNC_LDS 1
#else
#define USE_ASYNC_LDS 0
#endif

__device__ __forceinline__ void copy16_g2l(const half_t* g, half_t* l) {
#if USE_ASYNC_LDS
  __builtin_amdgcn_global_load_async_to_lds_b128(
      (gas_v4i)(void*)g, (las_v4i)(void*)l, 0, 0);
#else
  *reinterpret_cast<uint4*>(l) = *reinterpret_cast<const uint4*>(g);
#endif
}

__device__ __forceinline__ void lds_stage_fence() {
#if USE_ASYNC_LDS
  __builtin_amdgcn_s_wait_asynccnt(0);
#endif
  __syncthreads();
}

// ---------------- f32 -> f16 convert ----------------
__global__ void cvt_f32_to_f16(const float* __restrict__ x, half_t* __restrict__ y, int n) {
  int i = blockIdx.x * 256 + threadIdx.x;
  if (i < n) y[i] = (half_t)x[i];
}

// ---------------- f16 transpose: per batch z, in[rows,cols] -> out[cols,rows] ----------------
__global__ __launch_bounds__(256)
void transpose_f16(const half_t* __restrict__ in, half_t* __restrict__ out,
                   int rows, int cols) {
  __shared__ half_t tile[32][33];
  size_t zoff = (size_t)blockIdx.z * rows * cols;
  in += zoff; out += zoff;
  int r0 = blockIdx.y * 32, c0 = blockIdx.x * 32;
  int tx = threadIdx.x & 31, ty = threadIdx.x >> 5;   // 8 rows per pass
  for (int i = ty; i < 32; i += 8)
    tile[i][tx] = in[(size_t)(r0 + i) * cols + c0 + tx];
  __syncthreads();
  for (int i = ty; i < 32; i += 8)
    out[(size_t)(c0 + i) * rows + r0 + tx] = tile[tx][i];
}

// ---------------- LayerNorm (f32 in, f16 out) ----------------
__global__ __launch_bounds__(256)
void layernorm_f16(const float* __restrict__ x, const float* __restrict__ w,
                   const float* __restrict__ b, half_t* __restrict__ out, int d) {
  int row = blockIdx.x;
  const float* xr = x + (size_t)row * d;
  half_t* orow = out + (size_t)row * d;
  int tid = threadIdx.x;
  float s = 0.f, s2 = 0.f;
  for (int i = tid; i < d; i += 256) { float v = xr[i]; s += v; s2 += v * v; }
  for (int off = 16; off; off >>= 1) { s += __shfl_xor(s, off, 32); s2 += __shfl_xor(s2, off, 32); }
  __shared__ float r1[8], r2[8];
  if ((tid & 31) == 0) { r1[tid >> 5] = s; r2[tid >> 5] = s2; }
  __syncthreads();
  if (tid == 0) {
    float t1 = 0.f, t2 = 0.f;
    for (int i = 0; i < 8; ++i) { t1 += r1[i]; t2 += r2[i]; }
    r1[0] = t1; r2[0] = t2;
  }
  __syncthreads();
  float mean = r1[0] / d;
  float var  = r2[0] / d - mean * mean;
  float rstd = rsqrtf(var + 1e-5f);
  for (int i = tid; i < d; i += 256)
    orow[i] = (half_t)((xr[i] - mean) * rstd * w[i] + b[i]);
}

// ---------------- causal softmax: f32 scores -> f16 probs (zero-padded) ----------------
__global__ __launch_bounds__(256)
void softmax_causal(const float* __restrict__ scores, half_t* __restrict__ probs,
                    int s, float scale) {
  int q = blockIdx.x;
  const float* sr = scores + (size_t)q * s;
  half_t* pr = probs + (size_t)q * s;
  int L = q + 1;
  int tid = threadIdx.x;
  __shared__ float red[8];
  float mx = -3.0e38f;
  for (int i = tid; i < L; i += 256) mx = fmaxf(mx, sr[i] * scale);
  for (int off = 16; off; off >>= 1) mx = fmaxf(mx, __shfl_xor(mx, off, 32));
  if ((tid & 31) == 0) red[tid >> 5] = mx;
  __syncthreads();
  if (tid == 0) {
    float m = red[0];
    for (int i = 1; i < 8; ++i) m = fmaxf(m, red[i]);
    red[0] = m;
  }
  __syncthreads();
  mx = red[0];
  __syncthreads();
  float sum = 0.f;
  for (int i = tid; i < L; i += 256) {
    float e = __expf(sr[i] * scale - mx);
    pr[i] = (half_t)e;
    sum += e;
  }
  for (int off = 16; off; off >>= 1) sum += __shfl_xor(sum, off, 32);
  if ((tid & 31) == 0) red[tid >> 5] = sum;
  __syncthreads();
  if (tid == 0) {
    float t = 0.f;
    for (int i = 0; i < 8; ++i) t += red[i];
    red[0] = t;
  }
  __syncthreads();
  float inv = 1.0f / red[0];
  for (int i = tid; i < L; i += 256) pr[i] = (half_t)((float)pr[i] * inv);
  for (int i = L + tid; i < s; i += 256) pr[i] = (half_t)0.f;
}

// ---------------- WMMA GEMM ----------------
// C[M,N] = A[M,K] * B^T (+ epilogue); A f16 row-major [M,K], B f16 row-major [N,K].
// Block: 128 threads = 4 waves; tile 128x64; K chunk 32.
// Each wave: 32 rows x 64 cols = 2 A-frags x 4 B-frags = 8 WMMA per chunk.
#define TM 128
#define TN 64
#define TK 32
#define LDK 40   // padded K stride in LDS (halfs)

enum { EP_F16 = 0, EP_BIAS_F16 = 1, EP_F32 = 2, EP_BIAS_GELU_F16 = 3, EP_BIAS_RES_F32 = 4 };

__device__ __forceinline__ v16h load_frag(const half_t* base, int hi) {
  // 16-bit 16x32 matrix lane layout: lanes 0-15 hold K {0..7,16..23},
  // lanes 16-31 hold K {8..15,24..31}; K-contiguous LDS rows -> two b128 loads.
  union { v16h v; uint4 q[2]; } u;
  u.q[0] = *reinterpret_cast<const uint4*>(base + hi * 8);
  u.q[1] = *reinterpret_cast<const uint4*>(base + 16 + hi * 8);
  return u.v;
}

__global__ __launch_bounds__(128)
void gemm_wmma_f16(const half_t* __restrict__ A, const half_t* __restrict__ Bm,
                   half_t* __restrict__ Ch, float* __restrict__ Cf,
                   const float* __restrict__ bias, const float* __restrict__ res,
                   int M, int N, int K, int lda, int ldb, int ldc,
                   long long sA, long long sB, long long sC, long long sBias,
                   int ep, int causal) {
  int m0 = blockIdx.y * TM;
  int n0 = blockIdx.x * TN;
  if (causal && n0 > m0 + (TM - 1)) return;   // fully-masked score tile (uniform exit)
  int z = blockIdx.z;
  A  += (size_t)z * sA;
  Bm += (size_t)z * sB;
  if (Ch)   Ch   += (size_t)z * sC;
  if (Cf)   Cf   += (size_t)z * sC;
  if (bias) bias += (size_t)z * sBias;

  __shared__ __align__(16) half_t As[TM * LDK];   // [m][k]
  __shared__ __align__(16) half_t Bs[TN * LDK];   // [n][k]

  int tid  = threadIdx.x;
  int wave = tid >> 5;
  int lane = tid & 31;
  int hi   = lane >> 4;
  int l16  = lane & 15;
  bool fullB = (n0 + TN <= N);

  v8f acc[2][4];
  #pragma unroll
  for (int s = 0; s < 2; ++s)
    #pragma unroll
    for (int c = 0; c < 4; ++c)
      #pragma unroll
      for (int j = 0; j < 8; ++j) acc[s][c][j] = 0.0f;

  for (int k0 = 0; k0 < K; k0 += TK) {
    // A tile: 128x32 halfs = 512 b128 chunks (M,K multiples of 128/32 here)
    #pragma unroll
    for (int it = 0; it < 4; ++it) {
      int j  = tid + it * 128;
      int r  = j >> 2;
      int cg = (j & 3) * 8;
      copy16_g2l(A + (size_t)(m0 + r) * lda + k0 + cg, &As[r * LDK + cg]);
    }
    // B tile: 64x32 halfs = 256 b128 chunks; per-row guard for partial N tiles
    if (fullB) {
      #pragma unroll
      for (int it = 0; it < 2; ++it) {
        int j  = tid + it * 128;
        int n  = j >> 2;
        int cg = (j & 3) * 8;
        copy16_g2l(Bm + (size_t)(n0 + n) * ldb + k0 + cg, &Bs[n * LDK + cg]);
      }
    } else {
      #pragma unroll
      for (int it = 0; it < 2; ++it) {
        int j  = tid + it * 128;
        int n  = j >> 2;
        int cg = (j & 3) * 8;
        uint4 val = make_uint4(0u, 0u, 0u, 0u);
        if (n0 + n < N)
          val = *reinterpret_cast<const uint4*>(Bm + (size_t)(n0 + n) * ldb + k0 + cg);
        *reinterpret_cast<uint4*>(&Bs[n * LDK + cg]) = val;
      }
    }
    lds_stage_fence();

    v16h af0 = load_frag(&As[(wave * 32 + l16) * LDK], hi);
    v16h af1 = load_frag(&As[(wave * 32 + 16 + l16) * LDK], hi);
    #pragma unroll
    for (int cb = 0; cb < 4; ++cb) {
      v16h bf = load_frag(&Bs[(cb * 16 + l16) * LDK], hi);
      acc[0][cb] = __builtin_amdgcn_wmma_f32_16x16x32_f16(
          false, af0, false, bf, (short)0, acc[0][cb], false, false);
      acc[1][cb] = __builtin_amdgcn_wmma_f32_16x16x32_f16(
          false, af1, false, bf, (short)0, acc[1][cb], false, false);
    }
    __syncthreads();
  }

  // store: VGPR i, lanes 0-15 -> M=i, lanes 16-31 -> M=i+8; N = lane&15 within 16-col block
  #pragma unroll
  for (int sub = 0; sub < 2; ++sub) {
    #pragma unroll
    for (int cb = 0; cb < 4; ++cb) {
      int col = n0 + cb * 16 + l16;
      if (col >= N) continue;
      #pragma unroll
      for (int i = 0; i < 8; ++i) {
        int row = m0 + wave * 32 + sub * 16 + i + hi * 8;
        if (row >= M) continue;
        float v = acc[sub][cb][i];
        if (ep == EP_BIAS_F16 || ep == EP_BIAS_GELU_F16 || ep == EP_BIAS_RES_F32) v += bias[col];
        if (ep == EP_BIAS_GELU_F16) v = 0.5f * v * (1.0f + erff(v * 0.70710678118654752f));
        size_t idx = (size_t)row * ldc + col;
        if (ep == EP_BIAS_RES_F32) { v += res[idx]; Cf[idx] = v; }
        else if (ep == EP_F32)     { Cf[idx] = v; }
        else                       { Ch[idx] = (half_t)v; }
      }
    }
  }
}

// ---------------- host orchestration ----------------
static inline void launch_gemm(hipStream_t st,
                               const half_t* A, const half_t* B,
                               half_t* Ch, float* Cf,
                               const float* bias, const float* res,
                               int M, int N, int K, int lda, int ldb, int ldc,
                               long long sA, long long sB, long long sC, long long sBias,
                               int ep, int causal, int batches) {
  dim3 g((N + TN - 1) / TN, (M + TM - 1) / TM, batches);
  gemm_wmma_f16<<<g, 128, 0, st>>>(A, B, Ch, Cf, bias, res, M, N, K, lda, ldb, ldc,
                                   sA, sB, sC, sBias, ep, causal);
}

static inline void cvt(hipStream_t st, const float* x, half_t* y, int n) {
  cvt_f32_to_f16<<<(n + 255) / 256, 256, 0, st>>>(x, y, n);
}

extern "C" void kernel_launch(void* const* d_in, const int* in_sizes, int n_in,
                              void* d_out, int out_size, void* d_ws, size_t ws_size,
                              hipStream_t stream) {
  (void)in_sizes; (void)n_in; (void)out_size; (void)ws_size;
  const float* hs    = (const float*)d_in[0];
  const float* ln1w  = (const float*)d_in[1];
  const float* ln1b  = (const float*)d_in[2];
  const float* qU    = (const float*)d_in[3];
  const float* qV    = (const float*)d_in[4];
  const float* qb    = (const float*)d_in[5];
  const float* kU    = (const float*)d_in[6];
  const float* kV    = (const float*)d_in[7];
  const float* kb    = (const float*)d_in[8];
  const float* vU    = (const float*)d_in[9];
  const float* vV    = (const float*)d_in[10];
  const float* vb    = (const float*)d_in[11];
  const float* outU  = (const float*)d_in[12];
  const float* outV  = (const float*)d_in[13];
  const float* outb  = (const float*)d_in[14];
  const float* ln2w  = (const float*)d_in[15];
  const float* ln2b  = (const float*)d_in[16];
  const float* fc1U  = (const float*)d_in[17];
  const float* fc1V  = (const float*)d_in[18];
  const float* fc1b  = (const float*)d_in[19];
  const float* fc2U  = (const float*)d_in[20];
  const float* fc2V  = (const float*)d_in[21];
  const float* fc2b  = (const float*)d_in[22];
  float* out = (float*)d_out;

  // -------- workspace carve (256B aligned) --------
  char* p = (char*)d_ws;
  auto carve = [&](size_t bytes) -> char* {
    char* r = p;
    p += (bytes + 255) & ~(size_t)255;
    return r;
  };
  half_t* wN   = (half_t*)carve((size_t)N1 * DD * 2);            // normed (LN1 then LN2)
  half_t* wT   = (half_t*)carve((size_t)HH * N1 * R_ATTN * 2);   // per-head rank-32 temp
  half_t* wQ   = (half_t*)carve((size_t)HH * N1 * EE * 2);
  half_t* wK   = (half_t*)carve((size_t)HH * N1 * EE * 2);
  half_t* wV   = (half_t*)carve((size_t)HH * N1 * EE * 2);
  half_t* wVt  = (half_t*)carve((size_t)HH * EE * N1 * 2);       // V transposed per head
  float*  wSc  = (float*) carve((size_t)SS * SS * 4);            // scores (reused per b,h)
  half_t* wPr  = (half_t*)carve((size_t)SS * SS * 2);            // probs  (reused per b,h)
  half_t* wAt  = (half_t*)carve((size_t)N1 * DD * 2);            // concat attention
  half_t* wT2  = (half_t*)carve((size_t)N1 * R_OUT * 2);         // rank-512 temp (reused)
  half_t* wFF  = (half_t*)carve((size_t)N1 * DFF * 2);           // gelu(fc1)
  // f16 weights
  half_t* qUh  = (half_t*)carve((size_t)HH * EE * R_ATTN * 2);
  half_t* qVh  = (half_t*)carve((size_t)HH * R_ATTN * DD * 2);
  half_t* kUh  = (half_t*)carve((size_t)HH * EE * R_ATTN * 2);
  half_t* kVh  = (half_t*)carve((size_t)HH * R_ATTN * DD * 2);
  half_t* vUh  = (half_t*)carve((size_t)HH * EE * R_ATTN * 2);
  half_t* vVh  = (half_t*)carve((size_t)HH * R_ATTN * DD * 2);
  half_t* oUh  = (half_t*)carve((size_t)DD * R_OUT * 2);
  half_t* oVh  = (half_t*)carve((size_t)R_OUT * DD * 2);
  half_t* f1Uh = (half_t*)carve((size_t)DFF * R_FF * 2);
  half_t* f1Vh = (half_t*)carve((size_t)R_FF * DD * 2);
  half_t* f2Uh = (half_t*)carve((size_t)DD * R_FF * 2);
  half_t* f2Vh = (half_t*)carve((size_t)R_FF * DFF * 2);

  // -------- weight conversion --------
  cvt(stream, qU, qUh, HH * EE * R_ATTN);  cvt(stream, qV, qVh, HH * R_ATTN * DD);
  cvt(stream, kU, kUh, HH * EE * R_ATTN);  cvt(stream, kV, kVh, HH * R_ATTN * DD);
  cvt(stream, vU, vUh, HH * EE * R_ATTN);  cvt(stream, vV, vVh, HH * R_ATTN * DD);
  cvt(stream, outU, oUh, DD * R_OUT);      cvt(stream, outV, oVh, R_OUT * DD);
  cvt(stream, fc1U, f1Uh, DFF * R_FF);     cvt(stream, fc1V, f1Vh, R_FF * DD);
  cvt(stream, fc2U, f2Uh, DD * R_FF);      cvt(stream, fc2V, f2Vh, R_FF * DFF);

  // -------- LN1 --------
  layernorm_f16<<<N1, 256, 0, stream>>>(hs, ln1w, ln1b, wN, DD);

  // -------- QKV: t = normed * V_h^T (batched over heads), then * U_h^T + bias --------
  const long long sT = (long long)N1 * R_ATTN, sQ = (long long)N1 * EE;
  launch_gemm(stream, wN, qVh, wT, nullptr, nullptr, nullptr,
              N1, R_ATTN, DD, DD, DD, R_ATTN, 0, (long long)R_ATTN * DD, sT, 0, EP_F16, 0, HH);
  launch_gemm(stream, wT, qUh, wQ, nullptr, qb, nullptr,
              N1, EE, R_ATTN, R_ATTN, R_ATTN, EE, sT, (long long)EE * R_ATTN, sQ, EE, EP_BIAS_F16, 0, HH);
  launch_gemm(stream, wN, kVh, wT, nullptr, nullptr, nullptr,
              N1, R_ATTN, DD, DD, DD, R_ATTN, 0, (long long)R_ATTN * DD, sT, 0, EP_F16, 0, HH);
  launch_gemm(stream, wT, kUh, wK, nullptr, kb, nullptr,
              N1, EE, R_ATTN, R_ATTN, R_ATTN, EE, sT, (long long)EE * R_ATTN, sQ, EE, EP_BIAS_F16, 0, HH);
  launch_gemm(stream, wN, vVh, wT, nullptr, nullptr, nullptr,
              N1, R_ATTN, DD, DD, DD, R_ATTN, 0, (long long)R_ATTN * DD, sT, 0, EP_F16, 0, HH);
  launch_gemm(stream, wT, vUh, wV, nullptr, vb, nullptr,
              N1, EE, R_ATTN, R_ATTN, R_ATTN, EE, sT, (long long)EE * R_ATTN, sQ, EE, EP_BIAS_F16, 0, HH);

  // V^T per head: [N1,EE] -> [EE,N1], batched over H
  {
    dim3 g(EE / 32, N1 / 32, HH);
    transpose_f16<<<g, 256, 0, stream>>>(wV, wVt, N1, EE);
  }

  // -------- attention: per (b,h), reuse score/prob buffers --------
  for (int h = 0; h < HH; ++h) {
    for (int b = 0; b < BB; ++b) {
      const half_t* qp  = wQ  + ((size_t)h * N1 + (size_t)b * SS) * EE;
      const half_t* kp  = wK  + ((size_t)h * N1 + (size_t)b * SS) * EE;
      const half_t* vtp = wVt + (size_t)h * EE * N1 + (size_t)b * SS;   // [EE, SS] ldb=N1
      // scores = q * k^T (causal tiles only), f32
      launch_gemm(stream, qp, kp, nullptr, wSc, nullptr, nullptr,
                  SS, SS, EE, EE, EE, SS, 0, 0, 0, 0, EP_F32, 1, 1);
      softmax_causal<<<SS, 256, 0, stream>>>(wSc, wPr, SS, SCALE_QK);
      // attn = probs * (V^T)^T  -> concat layout [b*S+s, h*E+e]
      half_t* ap = wAt + (size_t)b * SS * DD + (size_t)h * EE;
      launch_gemm(stream, wPr, vtp, ap, nullptr, nullptr, nullptr,
                  SS, EE, SS, SS, N1, DD, 0, 0, 0, 0, EP_F16, 0, 1);
    }
  }

  // -------- out proj (rank 512) + residual -> out (f32) --------
  launch_gemm(stream, wAt, oVh, wT2, nullptr, nullptr, nullptr,
              N1, R_OUT, DD, DD, DD, R_OUT, 0, 0, 0, 0, EP_F16, 0, 1);
  launch_gemm(stream, wT2, oUh, nullptr, out, outb, hs,
              N1, DD, R_OUT, R_OUT, R_OUT, DD, 0, 0, 0, 0, EP_BIAS_RES_F32, 0, 1);

  // -------- LN2 --------
  layernorm_f16<<<N1, 256, 0, stream>>>(out, ln2w, ln2b, wN, DD);

  // -------- FFN: fc1 (rank 512) + gelu, fc2 (rank 512) + residual --------
  launch_gemm(stream, wN, f1Vh, wT2, nullptr, nullptr, nullptr,
              N1, R_FF, DD, DD, DD, R_FF, 0, 0, 0, 0, EP_F16, 0, 1);
  launch_gemm(stream, wT2, f1Uh, wFF, nullptr, fc1b, nullptr,
              N1, DFF, R_FF, R_FF, R_FF, DFF, 0, 0, 0, 0, EP_BIAS_GELU_F16, 0, 1);
  launch_gemm(stream, wFF, f2Vh, wT2, nullptr, nullptr, nullptr,
              N1, R_FF, DFF, DFF, DFF, R_FF, 0, 0, 0, 0, EP_F16, 0, 1);
  launch_gemm(stream, wT2, f2Uh, nullptr, out, fc2b, out,
              N1, DD, R_FF, R_FF, R_FF, DD, 0, 0, 0, 0, EP_BIAS_RES_F32, 0, 1);
}
